// WMSA_52183852646684
// MI455X (gfx1250) — matrix operand order repaired and checked by reading, not compiled, not run on hardware
//
#include <hip/hip_runtime.h>
#include <hip/hip_bf16.h>

// ---------------------------------------------------------------------------
// W-MSA (Swin window attention) for MI455X / gfx1250.
// bf16 WMMA everywhere + Tensor Data Mover (TDM) for LDS tile staging.
// ---------------------------------------------------------------------------

typedef __attribute__((ext_vector_type(16))) __bf16        v16bf;
typedef __attribute__((ext_vector_type(8)))  float         v8f;
typedef __attribute__((ext_vector_type(4)))  unsigned int  v4u;
typedef __attribute__((ext_vector_type(4)))  unsigned int  u32x4;
typedef __attribute__((ext_vector_type(8)))  int           i32x8;
typedef __attribute__((ext_vector_type(4)))  int           i32x4;

#define Bn   1024
#define Tn   49
#define Cn   256
#define Hn   8
#define Dn   32
#define BTn  (Bn * Tn)        // 50176 = 784 * 64 exactly
#define NPOS 13
#define QSCALE 0.17677669529663687f   // 1/sqrt(32)

// Native bf16 conversions (backend picks hardware cvt).
union BfPack { __bf16 h[2]; unsigned int u; };
__device__ __forceinline__ unsigned int pack2(float lo, float hi) {
  BfPack p; p.h[0] = (__bf16)lo; p.h[1] = (__bf16)hi; return p.u;
}
__device__ __forceinline__ void st_bf(unsigned short* p, float v) {
  *(__bf16*)p = (__bf16)v;
}

union Frag { unsigned int u[8]; v16bf v; };

// Load one 16-half fragment as two 16B LDS reads (ds_load_b128 x2).
__device__ __forceinline__ v16bf ld_frag(const unsigned int* lds, int d0, int d1) {
  Frag f;
  v4u a = *(const v4u*)(lds + d0);
  v4u b = *(const v4u*)(lds + d1);
  f.u[0] = a[0]; f.u[1] = a[1]; f.u[2] = a[2]; f.u[3] = a[3];
  f.u[4] = b[0]; f.u[5] = b[1]; f.u[6] = b[2]; f.u[7] = b[3];
  return f.v;
}

__device__ __forceinline__ v8f wmma_bf16(v16bf a, v16bf b, v8f c) {
  return __builtin_amdgcn_wmma_f32_16x16x32_bf16(
      false, a, false, b, (short)0, c, false, false);
}

// ---------------------------------------------------------------------------
// Tensor Data Mover: async 2D bf16 tile load  global -> LDS.
//   Tile: tile_d1 rows x tile_d0 elements (2B each), row stride = stride0 elems.
//   pad!=0 inserts 4 pad dwords after every 16 data dwords in LDS
//   (pad_interval code 3 = 16 dwords, pad_amount code 3 = 4 dwords),
//   matching our 32-half rows stored with a 20-dword LDS stride.
//   lds_byte_off is the byte offset of the destination inside the kernel's
//   single __shared__ block (placed at LDS offset 0).
// ---------------------------------------------------------------------------
__device__ __forceinline__ void tdm_load_2d(
    unsigned lds_byte_off, unsigned long long gaddr,
    unsigned tensor_d0, unsigned tensor_d1, unsigned long long stride0,
    unsigned tile_d0, unsigned tile_d1, int pad) {
  u32x4 g0;
  g0[0] = 1u;                                   // count=1, user descriptor
  g0[1] = lds_byte_off;                         // lds_addr
  g0[2] = (unsigned)(gaddr & 0xffffffffull);    // global_addr[31:0]
  g0[3] = (unsigned)((gaddr >> 32) & 0x01ffffffull) | 0x80000000u; // addr[56:32] | type=2
  i32x8 g1;
  g1[0] = (int)((1u << 16) |                    // data_size = 2 bytes
                (pad ? ((1u << 20) | (3u << 22) | (3u << 25)) : 0u));
  g1[1] = (int)((tensor_d0 & 0xffffu) << 16);                      // tensor_dim0[15:0]
  g1[2] = (int)(((tensor_d0 >> 16) & 0xffffu) |
                ((tensor_d1 & 0xffffu) << 16));                    // dim0 hi | dim1 lo
  g1[3] = (int)(((tensor_d1 >> 16) & 0xffffu) |
                ((tile_d0 & 0xffffu) << 16));                      // dim1 hi | tile_dim0
  g1[4] = (int)(tile_d1 & 0xffffu);                                // tile_dim1 (tile_dim2=0)
  g1[5] = (int)(unsigned)(stride0 & 0xffffffffull);                // tensor_dim0_stride lo
  g1[6] = (int)(unsigned)((stride0 >> 32) & 0xffffull);            // stride hi
  g1[7] = 0;
  i32x4 z4 = {0, 0, 0, 0};
  i32x8 z8 = {0, 0, 0, 0, 0, 0, 0, 0};
  __builtin_amdgcn_tensor_load_to_lds(g0, g1, z4, z4, z8, 0);
}

// ---------------------------------------------------------------------------
// Prep kernels
// ---------------------------------------------------------------------------

// Pack wq|wk|wv (each (C, H*D) f32) into B^T-major bf16: Wt[n][k], n in [0,768)
__global__ void prep_wqkv(const float* __restrict__ wq, const float* __restrict__ wk,
                          const float* __restrict__ wv, unsigned short* __restrict__ Wt) {
  int n = blockIdx.x, k = threadIdx.x;
  const float* w = (n < 256) ? wq : (n < 512) ? wk : wv;
  int nn = n & 255;
  st_bf(Wt + (size_t)n * 256 + k, w[(size_t)k * 256 + nn]);
}

// W2 = wo @ dw (256x256), stored B^T-major bf16: W2t[f][hd]
__global__ void prep_w2(const float* __restrict__ wo, const float* __restrict__ dw,
                        unsigned short* __restrict__ W2t) {
  int f = blockIdx.x, hd = threadIdx.x;
  float s = 0.f;
  for (int c = 0; c < 256; ++c)
    s += wo[(size_t)hd * 256 + c] * dw[(size_t)c * 256 + f];
  st_bf(W2t + (size_t)f * 256 + hd, s);
}

// b2 = bo @ dw + db
__global__ void prep_b2(const float* __restrict__ bo, const float* __restrict__ dw,
                        const float* __restrict__ db, float* __restrict__ b2) {
  int f = threadIdx.x;
  float s = db[f];
  for (int c = 0; c < 256; ++c) s += bo[c] * dw[(size_t)c * 256 + f];
  b2[f] = s;
}

// Dense gather of relative-position bias: biasg[h][q][k]
__global__ void prep_bias(const int* __restrict__ relpos, const float* __restrict__ btab,
                          float* __restrict__ biasg) {
  int idx = blockIdx.x * 256 + threadIdx.x;
  if (idx >= Hn * Tn * Tn) return;
  int h   = idx / (Tn * Tn);
  int rem = idx - h * (Tn * Tn);
  int r0 = relpos[rem];
  int r1 = relpos[Tn * Tn + rem];
  biasg[idx] = btab[((size_t)h * NPOS + r0) * NPOS + r1];
}

// ---------------------------------------------------------------------------
// Kernel 1: fused QKV GEMM  (50176 x 256) @ (256 x 768), bf16 WMMA
// grid (784, 12), block 256 (8 waves); each block computes a 64x64 tile.
// A (x, f32) converted to bf16 by threads; B (bf16 weights) staged by TDM.
// ---------------------------------------------------------------------------
__global__ __launch_bounds__(256) void qkv_gemm(
    const float* __restrict__ x, const unsigned short* __restrict__ Wt,
    const float* __restrict__ bq, const float* __restrict__ bk,
    const float* __restrict__ bv, unsigned short* __restrict__ qkv) {
  // smem: sA @0 (64x20 dw), sB @5120 (64x20 dw)
  __shared__ __attribute__((aligned(16))) unsigned char smem[2 * 5120];
  unsigned int* sA = (unsigned int*)smem;
  unsigned int* sB = (unsigned int*)(smem + 5120);

  int tid  = threadIdx.x;
  int m0   = blockIdx.x * 64;
  int n0   = blockIdx.y * 64;
  int w    = tid >> 5, lane = tid & 31;
  int mt   = w & 3;
  int nt0  = (w >> 2) << 1;
  int sm   = tid >> 2, kq = tid & 3;

  v8f c0 = {0,0,0,0,0,0,0,0}, c1 = {0,0,0,0,0,0,0,0};

  for (int ks = 0; ks < 8; ++ks) {
    if (ks) __syncthreads();
    // B tile (64 rows x 32 halfs) via Tensor Data Mover, wave 0 only.
    if (tid < 32) {
      unsigned long long gaddr =
          (unsigned long long)(size_t)Wt + ((size_t)n0 * 256 + ks * 32) * 2;
      tdm_load_2d(5120u, gaddr, 256u, 768u, 256ull, 32u, 64u, 1);
    }
    // A tile: convert x (f32) 64x32 -> bf16 in LDS (all threads).
    {
      const float* src = x + (size_t)(m0 + sm) * Cn + ks * 32 + kq * 8;
      float4 f0 = ((const float4*)src)[0];
      float4 f1 = ((const float4*)src)[1];
      v4u p;
      p[0] = pack2(f0.x, f0.y); p[1] = pack2(f0.z, f0.w);
      p[2] = pack2(f1.x, f1.y); p[3] = pack2(f1.z, f1.w);
      *(v4u*)(sA + sm * 20 + kq * 4) = p;
    }
    if (tid < 32) __builtin_amdgcn_s_wait_tensorcnt(0);
    __syncthreads();

    int ar   = mt * 16 + (lane & 15);
    int offA = (lane >> 4) << 3;        // lanes 16-31: +8 dwords (K=16..31)
    v16bf a  = ld_frag(sA, ar * 20 + offA, ar * 20 + offA + 4);
    int g8   = (lane >> 4) << 3;        // B: lanes 16-31 take K=16..31
    int nr   = nt0 * 16 + (lane & 15);
    v16bf b0 = ld_frag(sB, nr * 20 + g8,        nr * 20 + g8 + 4);
    v16bf b1 = ld_frag(sB, (nr + 16) * 20 + g8, (nr + 16) * 20 + g8 + 4);
    c0 = wmma_bf16(a, b0, c0);
    c1 = wmma_bf16(a, b1, c1);
  }

  // Epilogue: +bias, scale q by 1/sqrt(D), store bf16 at (b, which, h, t, d)
  int mbase = m0 + mt * 16 + ((lane >> 4) << 3);
  for (int j = 0; j < 2; ++j) {
    v8f acc = j ? c1 : c0;
    int n     = n0 + (nt0 + j) * 16 + (lane & 15);
    int which = n >> 8;
    int hd    = n & 255;
    float badd = (which == 0) ? bq[hd] : (which == 1) ? bk[hd] : bv[hd];
    float mul  = (which == 0) ? QSCALE : 1.0f;
    int h = (n >> 5) & 7, d = n & 31;
    int b_ = mbase / Tn, t = mbase - b_ * Tn;
    for (int r = 0; r < 8; ++r) {
      size_t idx = ((((size_t)b_ * 3 + which) * Hn + h) * Tn + t) * Dn + d;
      st_bf(qkv + idx, (acc[r] + badd) * mul);
      if (++t == Tn) { t = 0; ++b_; }
    }
  }
}

// ---------------------------------------------------------------------------
// Kernel 2: attention per (b, h).  T padded 49 -> 64.  4 waves / block.
// Q/K tiles staged by TDM; V transposed by threads; bf16 WMMA throughout.
// ---------------------------------------------------------------------------
__global__ __launch_bounds__(128) void attention(
    const unsigned short* __restrict__ qkv, const float* __restrict__ biasg,
    unsigned short* __restrict__ attnout) {
  // smem layout (bytes): sQ @0 (5120), sK @5120 (5120), sVt @10240 (4608),
  //                      sL @14848 (16640 f32), sAt @31488 (9216)
  __shared__ __attribute__((aligned(16))) unsigned char smem[40704];
  unsigned int* sQ  = (unsigned int*)smem;                    // 64 x 20 dw
  unsigned int* sK  = (unsigned int*)(smem + 5120);           // 64 x 20 dw
  unsigned int* sVt = (unsigned int*)(smem + 10240);          // 32 x 36 dw
  float*        sL  = (float*)(smem + 14848);                 // 64 x 65 f32
  unsigned int* sAt = (unsigned int*)(smem + 31488);          // 64 x 36 dw

  int tid = threadIdx.x;
  int bh  = blockIdx.x;
  int b_  = bh >> 3, h = bh & 7;

  size_t qoff = (((size_t)b_ * 3 + 0) * Hn + h) * (size_t)(Tn * Dn);
  size_t koff = (((size_t)b_ * 3 + 1) * Hn + h) * (size_t)(Tn * Dn);
  const unsigned short* vbase = qkv + (((size_t)b_ * 3 + 2) * Hn + h) * (size_t)(Tn * Dn);

  // Q and K: 49 rows x 32 halfs, TDM into padded LDS rows (stride 20 dw).
  // Rows 49..63 stay unwritten; all consumers mask them (see below).
  if (tid < 32) {
    unsigned long long base = (unsigned long long)(size_t)qkv;
    tdm_load_2d(0u,     base + qoff * 2, 32u, 49u, 32ull, 32u, 49u, 1);
    tdm_load_2d(5120u,  base + koff * 2, 32u, 49u, 32ull, 32u, 49u, 1);
  }
  // V^T (d x t), zero-padded, by all threads.
  for (int idx = tid; idx < 64 * 32; idx += 128) {
    int t = idx >> 5, d = idx & 31;
    unsigned short vv = (t < Tn) ? vbase[t * 32 + d] : (unsigned short)0;
    ((unsigned short*)sVt)[d * 72 + t] = vv;
  }
  if (tid < 32) __builtin_amdgcn_s_wait_tensorcnt(0);
  __syncthreads();

  int w = tid >> 5, lane = tid & 31;
  int mt = w;

  // logits = Q @ K^T (+ bias); 4 N-tiles per wave.
  {
    int ar   = mt * 16 + (lane & 15);
    int offA = (lane >> 4) << 3;
    v16bf a  = ld_frag(sQ, ar * 20 + offA, ar * 20 + offA + 4);
    int g8   = (lane >> 4) << 3;
    for (int nt = 0; nt < 4; ++nt) {
      int nr = nt * 16 + (lane & 15);
      v16bf bb = ld_frag(sK, nr * 20 + g8, nr * 20 + g8 + 4);
      v8f zero = {0,0,0,0,0,0,0,0};
      v8f c = wmma_bf16(a, bb, zero);
      int kk = nt * 16 + (lane & 15);
      for (int r = 0; r < 8; ++r) {
        int qr = mt * 16 + r + ((lane >> 4) << 3);
        float v;
        if (kk < Tn) {
          float bval = (qr < Tn) ? biasg[((size_t)h * Tn + qr) * Tn + kk] : 0.0f;
          v = c[r] + bval;
        } else {
          v = -1e30f;                    // mask padded keys (ignores garbage acc)
        }
        sL[qr * 65 + kk] = v;
      }
    }
  }
  __syncthreads();

  // Row softmax; re-quantize attn to bf16 into sAt. Rows 49..63 zeroed.
  if (tid < Tn) {
    float* row = &sL[tid * 65];
    float m = row[0];
    for (int kk = 1; kk < Tn; ++kk) m = fmaxf(m, row[kk]);
    float s = 0.f;
    for (int kk = 0; kk < Tn; ++kk) { float e = __expf(row[kk] - m); row[kk] = e; s += e; }
    float inv = 1.0f / s;
    unsigned short* ar = (unsigned short*)sAt + tid * 72;
    for (int kk = 0; kk < Tn; ++kk) st_bf(ar + kk, row[kk] * inv);
    for (int kk = Tn; kk < 64; ++kk) ar[kk] = 0;
  } else if (tid < 64) {
    unsigned short* ar = (unsigned short*)sAt + tid * 72;
    for (int kk = 0; kk < 64; ++kk) ar[kk] = 0;
  }
  __syncthreads();

  // out = attn(64x64) @ V(64x32): per wave one M-tile, both d-tiles, 2 K-steps.
  v8f o0 = {0,0,0,0,0,0,0,0}, o1 = {0,0,0,0,0,0,0,0};
  for (int s = 0; s < 2; ++s) {
    int ar   = mt * 16 + (lane & 15);
    int offA = (lane >> 4) << 3;
    int ab   = ar * 36 + s * 16 + offA;
    v16bf a  = ld_frag(sAt, ab, ab + 4);
    int g8   = (lane >> 4) << 3;
    int d0   = (lane & 15);
    int v0   = d0 * 36 + s * 16 + g8;
    int v1   = (d0 + 16) * 36 + s * 16 + g8;
    v16bf bb0 = ld_frag(sVt, v0, v0 + 4);
    v16bf bb1 = ld_frag(sVt, v1, v1 + 4);
    o0 = wmma_bf16(a, bb0, o0);
    o1 = wmma_bf16(a, bb1, o1);
  }

  for (int j = 0; j < 2; ++j) {
    v8f acc = j ? o1 : o0;
    int col = h * 32 + j * 16 + (lane & 15);
    for (int r = 0; r < 8; ++r) {
      int t = mt * 16 + r + ((lane >> 4) << 3);
      if (t < Tn)
        st_bf(attnout + ((size_t)b_ * Tn + t) * Cn + col, acc[r]);
    }
  }
}

// ---------------------------------------------------------------------------
// Kernel 3: fused out-proj + dense: (50176 x 256) @ W2(256x256) + b2 -> f32.
// Both A and B tiles staged by TDM (both already bf16 in global).
// ---------------------------------------------------------------------------
__global__ __launch_bounds__(256) void final_gemm(
    const unsigned short* __restrict__ A, const unsigned short* __restrict__ W2t,
    const float* __restrict__ b2, float* __restrict__ out) {
  __shared__ __attribute__((aligned(16))) unsigned char smem[2 * 5120];
  unsigned int* sA = (unsigned int*)smem;
  unsigned int* sB = (unsigned int*)(smem + 5120);

  int tid = threadIdx.x;
  int m0  = blockIdx.x * 64;
  int n0  = blockIdx.y * 64;
  int w   = tid >> 5, lane = tid & 31;
  int mt  = w & 3;
  int nt0 = (w >> 2) << 1;

  v8f c0 = {0,0,0,0,0,0,0,0}, c1 = {0,0,0,0,0,0,0,0};

  for (int ks = 0; ks < 8; ++ks) {
    if (ks) __syncthreads();
    if (tid < 32) {
      unsigned long long abase = (unsigned long long)(size_t)A;
      unsigned long long wbase = (unsigned long long)(size_t)W2t;
      tdm_load_2d(0u,    abase + ((size_t)m0 * 256 + ks * 32) * 2,
                  256u, (unsigned)BTn, 256ull, 32u, 64u, 1);
      tdm_load_2d(5120u, wbase + ((size_t)n0 * 256 + ks * 32) * 2,
                  256u, 256u, 256ull, 32u, 64u, 1);
      __builtin_amdgcn_s_wait_tensorcnt(0);
    }
    __syncthreads();

    int ar   = mt * 16 + (lane & 15);
    int offA = (lane >> 4) << 3;
    v16bf a  = ld_frag(sA, ar * 20 + offA, ar * 20 + offA + 4);
    int g8   = (lane >> 4) << 3;
    int nr   = nt0 * 16 + (lane & 15);
    v16bf b0 = ld_frag(sB, nr * 20 + g8,        nr * 20 + g8 + 4);
    v16bf b1 = ld_frag(sB, (nr + 16) * 20 + g8, (nr + 16) * 20 + g8 + 4);
    c0 = wmma_bf16(a, b0, c0);
    c1 = wmma_bf16(a, b1, c1);
  }

  int mbase = m0 + mt * 16 + ((lane >> 4) << 3);
  for (int j = 0; j < 2; ++j) {
    v8f acc = j ? c1 : c0;
    int n = n0 + (nt0 + j) * 16 + (lane & 15);
    float bias = b2[n];
    for (int r = 0; r < 8; ++r) {
      int M = mbase + r;
      out[(size_t)M * Cn + n] = acc[r] + bias;
    }
  }
}

// ---------------------------------------------------------------------------
// Launch
// ---------------------------------------------------------------------------
extern "C" void kernel_launch(void* const* d_in, const int* in_sizes, int n_in,
                              void* d_out, int out_size, void* d_ws, size_t ws_size,
                              hipStream_t stream) {
  const float* x  = (const float*)d_in[0];
  const int*   rp = (const int*)d_in[1];
  const float* bt = (const float*)d_in[2];
  const float* wq = (const float*)d_in[3];
  const float* bq = (const float*)d_in[4];
  const float* wk = (const float*)d_in[5];
  const float* bk = (const float*)d_in[6];
  const float* wv = (const float*)d_in[7];
  const float* bv = (const float*)d_in[8];
  const float* wo = (const float*)d_in[9];
  const float* bo = (const float*)d_in[10];
  const float* dw = (const float*)d_in[11];
  const float* db = (const float*)d_in[12];
  float* out = (float*)d_out;

  char* ws = (char*)d_ws;
  unsigned short* Wt    = (unsigned short*)(ws + 0);          // 768*256*2  = 393216
  unsigned short* W2t   = (unsigned short*)(ws + 393216);     // 256*256*2  = 131072
  float*          b2    = (float*)(ws + 524288);              // 1024
  float*          biasg = (float*)(ws + 525312);              // 8*49*49*4  = 76832
  unsigned short* qkv   = (unsigned short*)(ws + 602368);     // 77070336
  unsigned short* aout  = (unsigned short*)(ws + 77672704);   // 25690112
  (void)in_sizes; (void)n_in; (void)out_size; (void)ws_size;

  prep_wqkv<<<768, 256, 0, stream>>>(wq, wk, wv, Wt);
  prep_w2 <<<256, 256, 0, stream>>>(wo, dw, W2t);
  prep_b2 <<<1,   256, 0, stream>>>(bo, dw, db, b2);
  prep_bias<<<76, 256, 0, stream>>>(rp, bt, biasg);

  qkv_gemm <<<dim3(784, 12), 256, 0, stream>>>(x, Wt, bq, bk, bv, qkv);
  attention<<<8192,          128, 0, stream>>>(qkv, biasg, aout);
  final_gemm<<<dim3(784, 4), 256, 0, stream>>>(aout, W2t, b2, out);
}
